// AttnDecoderRNN_2972117369165
// MI455X (gfx1250) — matrix operand order repaired and checked by reading, not compiled, not run on hardware
//
#include <hip/hip_runtime.h>
#include <hip/hip_bf16.h>
#include <math.h>

// ---------------- problem constants ----------------
static constexpr int kB = 64;
static constexpr int kS = 256;
static constexpr int kH = 1024;
static constexpr int kE = 512;
static constexpr int kV = 10000;
static constexpr int kT = 64;
static constexpr int kD = 512;      // per-direction hidden
static constexpr int kG = 4 * kD;   // 2048 gate rows
static constexpr int kXH = kE + kH + kD; // 2048 = [emb|ctx|h_dir]

// ---------------- CDNA5 WMMA types ----------------
typedef __attribute__((ext_vector_type(16))) __bf16 v16bf;
typedef __attribute__((ext_vector_type(8)))  __bf16 v8bf;
typedef __attribute__((ext_vector_type(8)))  float  v8f;

union BF16x16 { v16bf v; v8bf h[2]; };

// =====================================================================
// Generic bf16 WMMA GEMM:  C[M,N] = A[M,K] @ W[N,K]^T (+ bias[N])
//   A, W row-major bf16; C row-major f32 with row stride ldc.
//   Block = 256 threads (8 waves); block tile 16(M) x 256(N);
//   each wave owns a 16x32 tile (two 16x16 WMMA accumulators sharing
//   one A fragment), K swept in steps of 32 via
//   v_wmma_f32_16x16x32_bf16 (f32 accumulate).
//   M must be a multiple of 16. N is handled by clamping OOB weight
//   rows to N-1 (their products land in lanes whose columns are never
//   stored) -- keeps the inner loop branch-free with EXEC all-ones.
// =====================================================================
__global__ __launch_bounds__(256)
void k_wmma_gemm(const __bf16* __restrict__ A, const __bf16* __restrict__ W,
                 const float* __restrict__ bias, float* __restrict__ C,
                 int M, int N, int K, long ldc)
{
    const int lane = threadIdx.x & 31;
    const int wave = threadIdx.x >> 5;
    const int m0 = blockIdx.x << 4;
    const int n0 = (blockIdx.y << 8) + (wave << 5);   // wave tile: 16x32

    // A fragment: lane<16 -> M=lane, K 0..15
    //             lane>=16 -> M=lane-16, K 16..31
    const int aM = m0 + (lane & 15);
    const int ak = (lane >> 4) << 4;   // 0 or 16
    // B fragment: lane<16 -> N=lane, K 0..15 ; lane>=16 -> N=lane-16, K 16..31
    const int bk = (lane >> 4) << 4;   // 0 or 16
    int nB0 = n0 + (lane & 15);
    int nB1 = nB0 + 16;
    if (nB0 > N - 1) nB0 = N - 1;      // clamp: OOB columns never stored
    if (nB1 > N - 1) nB1 = N - 1;

    const __bf16* aRow  = A + (size_t)aM * K;
    const __bf16* bRow0 = W + (size_t)nB0 * K;
    const __bf16* bRow1 = W + (size_t)nB1 * K;

    v8f acc0 = {}, acc1 = {};
    for (int k0 = 0; k0 < K; k0 += 32) {
        BF16x16 a, b0, b1;
        a.h[0]  = *(const v8bf*)(aRow + k0 + ak);
        a.h[1]  = *(const v8bf*)(aRow + k0 + ak + 8);
        b0.h[0] = *(const v8bf*)(bRow0 + k0 + bk);
        b0.h[1] = *(const v8bf*)(bRow0 + k0 + bk + 8);
        b1.h[0] = *(const v8bf*)(bRow1 + k0 + bk);
        b1.h[1] = *(const v8bf*)(bRow1 + k0 + bk + 8);
        // stage next weight K-slices toward L0 (global_prefetch_b8)
        __builtin_prefetch((const void*)(bRow0 + k0 + 32 + bk), 0, 1);
        __builtin_prefetch((const void*)(bRow1 + k0 + 32 + bk), 0, 1);
        acc0 = __builtin_amdgcn_wmma_f32_16x16x32_bf16(
                   false, a.v, false, b0.v, (short)0, acc0, false, false);
        acc1 = __builtin_amdgcn_wmma_f32_16x16x32_bf16(
                   false, a.v, false, b1.v, (short)0, acc1, false, false);
    }

    // C/D layout: VGPR r, lane<16 -> M=m0+r ; lane>=16 -> M=m0+8+r ; N=n0+(lane&15)
    const int mhi = (lane >> 4) << 3;
    const int cn0 = n0 + (lane & 15);
    const int cn1 = cn0 + 16;
    if (cn0 < N) {
        const float bv = bias ? bias[cn0] : 0.0f;
#pragma unroll
        for (int r = 0; r < 8; ++r)
            C[(size_t)(m0 + mhi + r) * (size_t)ldc + cn0] = acc0[r] + bv;
    }
    if (cn1 < N) {
        const float bv = bias ? bias[cn1] : 0.0f;
#pragma unroll
        for (int r = 0; r < 8; ++r)
            C[(size_t)(m0 + mhi + r) * (size_t)ldc + cn1] = acc1[r] + bv;
    }
}

// ---------------- elementwise / small kernels ----------------
__global__ void k_f32_to_bf16(const float* __restrict__ s, __bf16* __restrict__ d, int n) {
    int i = blockIdx.x * 256 + threadIdx.x;
    if (i < n) d[i] = (__bf16)s[i];
}

__global__ void k_copy_f32(const float* __restrict__ s, float* __restrict__ d, int n) {
    int i = blockIdx.x * 256 + threadIdx.x;
    if (i < n) d[i] = s[i];
}

// Wcat[row, 0:1536] = Wih[row], Wcat[row, 1536:2048] = Whh[row]   (one direction)
__global__ void k_build_wcat(const float* __restrict__ Wih, const float* __restrict__ Whh,
                             __bf16* __restrict__ dst) {
    int i = blockIdx.x * 256 + threadIdx.x;           // over 2048*2048
    int row = i >> 11, col = i & 2047;
    float v = (col < kE + kH) ? Wih[row * (kE + kH) + col]
                              : Whh[row * kD + (col - (kE + kH))];
    dst[i] = (__bf16)v;
}

__global__ void k_bias_sum(const float* __restrict__ a, const float* __restrict__ b,
                           float* __restrict__ o, int n) {
    int i = blockIdx.x * 256 + threadIdx.x;
    if (i < n) o[i] = a[i] + b[i];
}

// query = [h_fwd || h_bwd]  as bf16  [B, H]
__global__ void k_query(const float* __restrict__ hs, __bf16* __restrict__ q) {
    int i = blockIdx.x * 256 + threadIdx.x;           // over B*H
    int b = i >> 10, j = i & 1023;
    int dir = j >> 9, d = j & 511;
    q[i] = (__bf16)hs[((size_t)dir * kB + b) * kD + d];
}

// sc[b,s] = sum_h tanh(qWa[b,h] + Uk[b,s,h]) * Va[h]   (one wave32 per (b,s))
__global__ __launch_bounds__(256)
void k_scores(const float* __restrict__ qWa, const float* __restrict__ Uk,
              const float* __restrict__ Va, float* __restrict__ sc) {
    int lane = threadIdx.x & 31;
    int gw = blockIdx.x * 8 + (threadIdx.x >> 5);     // 0..B*S-1
    int b = gw >> 8;
    const float* uk = Uk + (size_t)gw * kH;
    const float* qa = qWa + (size_t)b * kH;
    float acc = 0.0f;
    for (int h = lane; h < kH; h += 32)
        acc += tanhf(qa[h] + uk[h]) * Va[h];
#pragma unroll
    for (int off = 16; off > 0; off >>= 1)
        acc += __shfl_xor(acc, off, 32);
    if (lane == 0) sc[gw] = acc;
}

// softmax over S per batch row; also writes attention output slice
__global__ __launch_bounds__(256)
void k_softmax(const float* __restrict__ sc, const unsigned char* __restrict__ mask,
               float* __restrict__ w, float* __restrict__ attn /* d_out+attn_off+t*S */) {
    __shared__ float red[kS];
    int b = blockIdx.x, s = threadIdx.x;
    float v = sc[b * kS + s];
    if (mask[b * kS + s]) v = -INFINITY;
    red[s] = v; __syncthreads();
    for (int off = 128; off > 0; off >>= 1) {
        if (s < off) red[s] = fmaxf(red[s], red[s + off]);
        __syncthreads();
    }
    float mx = red[0]; __syncthreads();
    float e = __expf(v - mx);
    red[s] = e; __syncthreads();
    for (int off = 128; off > 0; off >>= 1) {
        if (s < off) red[s] += red[s + off];
        __syncthreads();
    }
    float wv = e / red[0];
    w[b * kS + s] = wv;
    attn[(size_t)b * kT * kS + s] = wv;
}

// ctx[b,h] = sum_s w[b,s] * enc[b,s,h]
__global__ void k_ctx(const float* __restrict__ w, const float* __restrict__ enc,
                      float* __restrict__ ctx) {
    int i = blockIdx.x * 256 + threadIdx.x;           // over B*H
    int b = i >> 10, h = i & 1023;
    const float* wr = w + b * kS;
    const float* er = enc + ((size_t)b * kS) * kH + h;
    float a = 0.0f;
    for (int s = 0; s < kS; ++s) a = fmaf(wr[s], er[(size_t)s * kH], a);
    ctx[i] = a;
}

// xh[dir][b] = bf16([ emb[tok] | ctx[b] | h_state[dir][b] ])   [2,B,2048]
__global__ void k_buildx(const int* __restrict__ target, const int* __restrict__ sosp,
                         const float* __restrict__ emb, const float* __restrict__ ctx,
                         const float* __restrict__ hs, __bf16* __restrict__ xh, int t) {
    int i = blockIdx.x * 256 + threadIdx.x;           // over 2*B*2048
    int dir = i >> 17, rem = i & 131071;
    int b = rem >> 11, j = rem & 2047;
    float v;
    if (j < kE) {
        int tok = (t == 0) ? sosp[0] : target[b * kT + (t - 1)];
        v = emb[(size_t)tok * kE + j];
    } else if (j < kE + kH) {
        v = ctx[b * kH + (j - kE)];
    } else {
        v = hs[((size_t)dir * kB + b) * kD + (j - kE - kH)];
    }
    xh[i] = (__bf16)v;
}

// LSTM activations: gate order i,f,g,o; update h,c; pack out = [hf||hb] bf16
__global__ void k_lstm(const float* __restrict__ g, float* __restrict__ hs,
                       float* __restrict__ cs, __bf16* __restrict__ out_bf) {
    int i = blockIdx.x * 256 + threadIdx.x;           // over 2*B*D
    int dir = i >> 15, rem = i & 32767;
    int b = rem >> 9, d = rem & 511;
    size_t base = ((size_t)dir * kB + b) * kG;
    float gi = g[base + d];
    float gf = g[base + kD + d];
    float gc = g[base + 2 * kD + d];
    float go = g[base + 3 * kD + d];
    float si = 1.0f / (1.0f + __expf(-gi));
    float sf = 1.0f / (1.0f + __expf(-gf));
    float so = 1.0f / (1.0f + __expf(-go));
    float c2 = sf * cs[i] + si * tanhf(gc);
    float h2 = so * tanhf(c2);
    cs[i] = c2;
    hs[i] = h2;
    out_bf[(size_t)b * kH + dir * kD + d] = (__bf16)h2;
}

__global__ void k_final(const float* __restrict__ hs, const float* __restrict__ cs,
                        float* __restrict__ oh, float* __restrict__ oc) {
    int i = blockIdx.x * 256 + threadIdx.x;           // over 2*B*D
    if (i < 2 * kB * kD) { oh[i] = hs[i]; oc[i] = cs[i]; }
}

// =====================================================================
extern "C" void kernel_launch(void* const* d_in, const int* in_sizes, int n_in,
                              void* d_out, int out_size, void* d_ws, size_t ws_size,
                              hipStream_t stream) {
    (void)in_sizes; (void)n_in; (void)out_size; (void)ws_size;

    const float* enc    = (const float*)d_in[0];
    const float* h0     = (const float*)d_in[1];
    const float* c0     = (const float*)d_in[2];
    const int*   target = (const int*)d_in[3];
    const unsigned char* mask = (const unsigned char*)d_in[4];
    const int*   sosp   = (const int*)d_in[5];
    const float* emb    = (const float*)d_in[6];
    const float* Wa     = (const float*)d_in[7];
    const float* Ua     = (const float*)d_in[8];
    const float* Va     = (const float*)d_in[9];
    const float* outW   = (const float*)d_in[10];
    const float* outb   = (const float*)d_in[11];
    const float* Wih_f  = (const float*)d_in[12];
    const float* Whh_f  = (const float*)d_in[13];
    const float* bih_f  = (const float*)d_in[14];
    const float* bhh_f  = (const float*)d_in[15];
    const float* Wih_b  = (const float*)d_in[16];
    const float* Whh_b  = (const float*)d_in[17];
    const float* bih_b  = (const float*)d_in[18];
    const float* bhh_b  = (const float*)d_in[19];

    float* out = (float*)d_out;
    const size_t H_OFF    = (size_t)kB * kT * kV;         // decoder_outputs
    const size_t C_OFF    = H_OFF + (size_t)2 * kB * kD;  // h
    const size_t ATTN_OFF = C_OFF + (size_t)2 * kB * kD;  // c -> attentions

    // ------- workspace carve-out (256B aligned) -------
    char* p = (char*)d_ws;
    auto carve = [&](size_t bytes) { char* r = p; p += (bytes + 255) & ~(size_t)255; return r; };
    float*  Uk       = (float*)  carve((size_t)kB * kS * kH * 4);   // 64 MB
    __bf16* enc_bf   = (__bf16*) carve((size_t)kB * kS * kH * 2);   // 32 MB
    __bf16* Ua_bf    = (__bf16*) carve((size_t)kH * kH * 2);
    __bf16* Wa_bf    = (__bf16*) carve((size_t)kH * kH * 2);
    __bf16* outW_bf  = (__bf16*) carve((size_t)kV * kH * 2);        // 20 MB
    __bf16* Wcat     = (__bf16*) carve((size_t)2 * kG * kXH * 2);   // 16 MB
    float*  bsum     = (float*)  carve((size_t)2 * kG * 4);
    float*  h_state  = (float*)  carve((size_t)2 * kB * kD * 4);
    float*  c_state  = (float*)  carve((size_t)2 * kB * kD * 4);
    __bf16* query_bf = (__bf16*) carve((size_t)kB * kH * 2);
    float*  qWa      = (float*)  carve((size_t)kB * kH * 4);
    float*  scores   = (float*)  carve((size_t)kB * kS * 4);
    float*  wgt      = (float*)  carve((size_t)kB * kS * 4);
    float*  ctxb     = (float*)  carve((size_t)kB * kH * 4);
    __bf16* xh       = (__bf16*) carve((size_t)2 * kB * kXH * 2);
    float*  gbuf     = (float*)  carve((size_t)2 * kB * kG * 4);
    __bf16* out_bf   = (__bf16*) carve((size_t)kB * kH * 2);

    // ------- one-time precompute -------
    k_f32_to_bf16<<<(kB*kS*kH + 255)/256, 256, 0, stream>>>(enc, enc_bf, kB*kS*kH);
    k_f32_to_bf16<<<(kH*kH + 255)/256,    256, 0, stream>>>(Ua, Ua_bf, kH*kH);
    k_f32_to_bf16<<<(kH*kH + 255)/256,    256, 0, stream>>>(Wa, Wa_bf, kH*kH);
    k_f32_to_bf16<<<(kV*kH + 255)/256,    256, 0, stream>>>(outW, outW_bf, kV*kH);
    k_build_wcat<<<(kG*kXH)/256, 256, 0, stream>>>(Wih_f, Whh_f, Wcat);
    k_build_wcat<<<(kG*kXH)/256, 256, 0, stream>>>(Wih_b, Whh_b, Wcat + (size_t)kG*kXH);
    k_bias_sum<<<(kG + 255)/256, 256, 0, stream>>>(bih_f, bhh_f, bsum, kG);
    k_bias_sum<<<(kG + 255)/256, 256, 0, stream>>>(bih_b, bhh_b, bsum + kG, kG);
    k_copy_f32<<<(2*kB*kD + 255)/256, 256, 0, stream>>>(h0, h_state, 2*kB*kD);
    k_copy_f32<<<(2*kB*kD + 255)/256, 256, 0, stream>>>(c0, c_state, 2*kB*kD);

    // Uk = enc[16384,1024] @ Ua^T   (L2-resident thereafter)
    k_wmma_gemm<<<dim3((kB*kS)/16, kH/256), 256, 0, stream>>>(
        enc_bf, Ua_bf, nullptr, Uk, kB*kS, kH, kH, (long)kH);

    // ------- sequential decode loop -------
    for (int t = 0; t < kT; ++t) {
        k_query<<<(kB*kH)/256, 256, 0, stream>>>(h_state, query_bf);

        k_wmma_gemm<<<dim3(kB/16, kH/256), 256, 0, stream>>>(
            query_bf, Wa_bf, nullptr, qWa, kB, kH, kH, (long)kH);

        k_scores<<<(kB*kS)/8, 256, 0, stream>>>(qWa, Uk, Va, scores);

        k_softmax<<<kB, kS, 0, stream>>>(
            scores, mask, wgt, out + ATTN_OFF + (size_t)t * kS);

        k_ctx<<<(kB*kH)/256, 256, 0, stream>>>(wgt, enc, ctxb);

        k_buildx<<<(2*kB*kXH)/256, 256, 0, stream>>>(
            target, sosp, emb, ctxb, h_state, xh, t);

        // gates, both directions: [64,2048] @ [2048,2048]^T + bias
        k_wmma_gemm<<<dim3(kB/16, kG/256), 256, 0, stream>>>(
            xh, Wcat, bsum, gbuf, kB, kG, kXH, (long)kG);
        k_wmma_gemm<<<dim3(kB/16, kG/256), 256, 0, stream>>>(
            xh + (size_t)kB*kXH, Wcat + (size_t)kG*kXH, bsum + kG,
            gbuf + (size_t)kB*kG, kB, kG, kXH, (long)kG);

        k_lstm<<<(2*kB*kD)/256, 256, 0, stream>>>(gbuf, h_state, c_state, out_bf);

        // logits straight into d_out[b,t,:] (ldc = T*V)
        k_wmma_gemm<<<dim3(kB/16, (kV + 255)/256), 256, 0, stream>>>(
            out_bf, outW_bf, outb, out + (size_t)t * kV, kB, kV, kH, (long)kT * kV);
    }

    k_final<<<(2*kB*kD + 255)/256, 256, 0, stream>>>(
        h_state, c_state, out + H_OFF, out + C_OFF);
}
